// PiGNNLayer_88046829568258
// MI455X (gfx1250) — compile-verified, matrix operand order checked by reading, not run on hardware
//
#include <hip/hip_runtime.h>
#include <hip/hip_bf16.h>
#include <math.h>

typedef __attribute__((ext_vector_type(16))) _Float16 v16h;
typedef __attribute__((ext_vector_type(8)))  _Float16 v8h;
typedef __attribute__((ext_vector_type(4)))  _Float16 v4h;
typedef __attribute__((ext_vector_type(8)))  float    v8f;
typedef __attribute__((ext_vector_type(4)))  float    v4f;
typedef __attribute__((ext_vector_type(4)))  unsigned int v4u;
typedef __attribute__((ext_vector_type(8)))  int      v8i;
typedef __attribute__((ext_vector_type(4)))  int      v4i;

#define NNODE 16384
#define KNBR  30
#define DEMB  128
#define NEDGE (NNODE*KNBR)
#define NTILE (NEDGE/16)
#define AST   136   // LDS activation row stride in halfs (16B-aligned rows, bank-spread)
#define TPW   8     // tiles per wave
#define WAVES 8     // waves per block

// ---- workspace layout ----
// half-element offsets for weight fragments (contiguous at start of ws; TDM copies whole region)
constexpr size_t HW_AW1 = 0;                       // 384x128
constexpr size_t HW_AW2 = HW_AW1 + 384*128;        // 128x128
constexpr size_t HW_AW3 = HW_AW2 + 128*128;        // 128x16 (padded from 128x1)
constexpr size_t HW_NW1 = HW_AW3 + 128*16;         // 256x128
constexpr size_t HW_NW2 = HW_NW1 + 256*128;        // 128x128
constexpr size_t HW_NW3 = HW_NW2 + 128*128;        // 128x128
constexpr size_t HW_THW = HW_NW3 + 128*128;        // 128x128 (used by to_h kernel, not staged)
constexpr size_t HW_END = HW_THW + 128*128;
constexpr size_t EDGE_W_HALFS = HW_THW;            // halfs staged into LDS by edge kernel
constexpr size_t EDGE_W_BYTES = EDGE_W_HALFS * 2;  // 266,240 B (AW1..NW3)
// byte offsets
constexpr size_t WS_LOG = HW_END * 2;                          // E fp32 logits
constexpr size_t WS_V   = WS_LOG + (size_t)NEDGE * 4;          // E x 128 f16 values
constexpr size_t WS_AGG = WS_V   + (size_t)NEDGE * DEMB * 2;   // N x 128 f16 agg
constexpr size_t WS_HH  = WS_AGG + (size_t)NNODE * DEMB * 2;   // N x 128 fp32 hh
constexpr size_t WS_COL = WS_HH  + (size_t)NNODE * DEMB * 4;   // 128 fp32 colsum
constexpr size_t WS_G   = WS_COL + 512;                        // 128 fp32 gate

// ---------------- device helpers ----------------
__device__ __forceinline__ v8f wmma16(v16h a, v16h b, v8f c) {
    return __builtin_amdgcn_wmma_f32_16x16x32_f16(false, a, false, b, (short)0, c, false, false);
}

// B fragment: 16 contiguous halfs per lane (prepacked); reads become ds_load_b128 from LDS
__device__ __forceinline__ v16h b_frag(const _Float16* __restrict__ p) {
    return *(const v16h*)p;
}

// A fragment from an fp32 row: runs [k0..k0+7] and [k0+16..k0+23] (p points at col k0)
__device__ __forceinline__ v16h a_frag_f32(const float* __restrict__ p) {
    v4f f0 = *(const v4f*)(p);
    v4f f1 = *(const v4f*)(p + 4);
    v4f f2 = *(const v4f*)(p + 16);
    v4f f3 = *(const v4f*)(p + 20);
    v16h a;
#pragma unroll
    for (int i = 0; i < 4; ++i) {
        a[i]      = (_Float16)f0[i];
        a[i + 4]  = (_Float16)f1[i];
        a[i + 8]  = (_Float16)f2[i];
        a[i + 12] = (_Float16)f3[i];
    }
    return a;
}

// A fragment from an f16 row (LDS or global), same run pattern, 8B-aligned chunks
__device__ __forceinline__ v16h a_frag_f16(const _Float16* __restrict__ p) {
    v4h h0 = *(const v4h*)(p);
    v4h h1 = *(const v4h*)(p + 4);
    v4h h2 = *(const v4h*)(p + 16);
    v4h h3 = *(const v4h*)(p + 20);
    v16h a;
#pragma unroll
    for (int i = 0; i < 4; ++i) {
        a[i]      = h0[i];
        a[i + 4]  = h1[i];
        a[i + 8]  = h2[i];
        a[i + 12] = h3[i];
    }
    return a;
}

__device__ __forceinline__ float gelu_exact(float x) {
    return 0.5f * x * (1.0f + erff(x * 0.70710678118654752f));
}

__device__ __forceinline__ void lds_fence() {
    asm volatile("s_wait_dscnt 0" ::: "memory");
}

// ---------------- weight repack: fp32 row-major [K x Nout] -> f16 B-fragments ----------------
__global__ void pignn_repack_kernel(const float* __restrict__ W, _Float16* __restrict__ dst,
                                    int Kdim, int Nout, int Npad) {
    const int NT = Npad >> 4;
    const int total = Kdim * Npad;
    for (int i = blockIdx.x * blockDim.x + threadIdx.x; i < total;
         i += gridDim.x * blockDim.x) {
        const int frag = i >> 9;
        const int rem  = i & 511;
        const int lane = rem >> 4;
        const int j    = rem & 15;
        const int kt = frag / NT, nt = frag - kt * NT;
        const int khalf = lane >> 4, n16 = lane & 15;
        const int kk = (j < 8) ? (khalf * 8 + j) : (khalf * 8 + j + 8); // j>=8 -> 16 + khalf*8 + (j-8)
        const int k = kt * 32 + kk;
        const int n = nt * 16 + n16;
        const float v = (n < Nout) ? W[(size_t)k * Nout + n] : 0.0f;
        dst[i] = (_Float16)v;
    }
}

__global__ void pignn_zero_kernel(float* __restrict__ p, int n) {
    const int i = blockIdx.x * blockDim.x + threadIdx.x;
    if (i < n) p[i] = 0.0f;
}

// ---------------- edge kernel: both edge MLPs, WMMA f16, weights TDM-staged in LDS ----------------
__global__ void __launch_bounds__(256)
pignn_edge_kernel(const float* __restrict__ h, const float* __restrict__ e,
                  const float* __restrict__ ab1, const float* __restrict__ ab2,
                  const float* __restrict__ ab3,
                  const float* __restrict__ nb1, const float* __restrict__ nb2,
                  const float* __restrict__ nb3,
                  const int* __restrict__ eidx,
                  const _Float16* __restrict__ wsW,
                  float* __restrict__ wsLog, _Float16* __restrict__ wsV) {
    // all six edge-MLP weight matrices as WMMA B-fragments (266,240 B) + per-wave act tiles
    __shared__ __align__(64) _Float16 sW[EDGE_W_HALFS];
    __shared__ __align__(64) _Float16 sAct[WAVES][16 * AST];

    const int wave  = threadIdx.x >> 5;
    const int lane  = threadIdx.x & 31;
    const int m     = lane & 15;
    const int khalf = lane >> 4;
    const int ncol  = lane & 15;

    // ---- Tensor Data Mover: DMA the whole fragment region ws[0 .. EDGE_W_BYTES) into LDS ----
    if (wave == 0) {
        const unsigned lds_off = (unsigned)(size_t)(&sW[0]);   // LDS aperture: low 32 bits = LDS byte offset
        const unsigned long long ga = (unsigned long long)(size_t)wsW;
        const unsigned n64 = (unsigned)(EDGE_W_BYTES / 8);     // elements of 8 bytes
        v4u g0;
        g0[0] = 1u;                                            // count=1, user descriptor
        g0[1] = lds_off;                                       // lds_addr
        g0[2] = (unsigned)(ga & 0xFFFFFFFFu);                  // global_addr[31:0]
        g0[3] = (unsigned)((ga >> 32) & 0x01FFFFFFu) | (2u << 30); // global_addr[56:32] | type=2
        v8i g1;
        g1[0] = (int)(3u << 16);                               // workgroup_mask=0, data_size=8B
        g1[1] = (int)((n64 & 0xFFFFu) << 16);                  // tensor_dim0[15:0] @ bits63:48
        g1[2] = (int)(((n64 >> 16) & 0xFFFFu) | (1u << 16));   // tensor_dim0[31:16] | tensor_dim1=1
        g1[3] = (int)((n64 & 0xFFFFu) << 16);                  // tile_dim0 @ bits127:112
        g1[4] = 0;                                             // tile_dim1=0, tile_dim2=0 (1-D)
        g1[5] = (int)n64;                                      // tensor_dim0_stride[31:0]
        g1[6] = 0;                                             // stride[47:32], dim1_stride lo
        g1[7] = 0;
        const v4i gz4 = {0, 0, 0, 0};
        const v8i gz8 = {0, 0, 0, 0, 0, 0, 0, 0};
        __builtin_amdgcn_tensor_load_to_lds(g0, g1, gz4, gz4, gz8, 0);
        __builtin_amdgcn_s_wait_tensorcnt(0);
    }
    __syncthreads();   // publish staged weights to all 8 waves

    _Float16* act = sAct[wave];
    const v8f vzero = {0.f, 0.f, 0.f, 0.f, 0.f, 0.f, 0.f, 0.f};
    const int tile0 = (blockIdx.x * WAVES + wave) * TPW;

    for (int it = 0; it < TPW; ++it) {
        const int tile = tile0 + it;
        const int edge_m = tile * 16 + m;
        const int src = eidx[edge_m];
        const int dst = eidx[NEDGE + edge_m];
        const float* hi = h + (size_t)src * DEMB;
        const float* ej = e + (size_t)edge_m * DEMB;
        const float* hj = h + (size_t)dst * DEMB;

        // speculative prefetch of next tile's e row
        __builtin_prefetch(ej + 16 * DEMB, 0, 1);

        // ===== att_mlp layer 1: cat3[16,384] @ aw1[384,128], ReLU =====
        v8f acc[8];
#pragma unroll
        for (int nt = 0; nt < 8; ++nt) acc[nt] = vzero;
#pragma unroll
        for (int kt = 0; kt < 12; ++kt) {
            const int c0 = kt * 32 + khalf * 8;
            const int f  = kt >> 2;                       // 0:hi 1:e 2:hj
            const float* rp = (f == 0 ? hi : (f == 1 ? ej : hj)) + (c0 - f * 128);
            const v16h a = a_frag_f32(rp);
#pragma unroll
            for (int nt = 0; nt < 8; ++nt) {
                const v16h b = b_frag(sW + HW_AW1 + ((size_t)(kt * 8 + nt) * 32 + lane) * 16);
                acc[nt] = wmma16(a, b, acc[nt]);
            }
        }
#pragma unroll
        for (int nt = 0; nt < 8; ++nt) {
            const float bias = ab1[nt * 16 + ncol];
#pragma unroll
            for (int v = 0; v < 8; ++v) {
                float x = acc[nt][v] + bias;
                x = x > 0.f ? x : 0.f;
                act[(khalf * 8 + v) * AST + nt * 16 + ncol] = (_Float16)x;
            }
        }
        lds_fence();

        // ===== att_mlp layer 2: [16,128] @ aw2[128,128], ReLU =====
#pragma unroll
        for (int nt = 0; nt < 8; ++nt) acc[nt] = vzero;
#pragma unroll
        for (int kt = 0; kt < 4; ++kt) {
            const v16h a = a_frag_f16(act + m * AST + kt * 32 + khalf * 8);
#pragma unroll
            for (int nt = 0; nt < 8; ++nt) {
                const v16h b = b_frag(sW + HW_AW2 + ((size_t)(kt * 8 + nt) * 32 + lane) * 16);
                acc[nt] = wmma16(a, b, acc[nt]);
            }
        }
#pragma unroll
        for (int nt = 0; nt < 8; ++nt) {
            const float bias = ab2[nt * 16 + ncol];
#pragma unroll
            for (int v = 0; v < 8; ++v) {
                float x = acc[nt][v] + bias;
                x = x > 0.f ? x : 0.f;
                act[(khalf * 8 + v) * AST + nt * 16 + ncol] = (_Float16)x;
            }
        }
        lds_fence();

        // ===== att_mlp layer 3: [16,128] @ aw3[128,16pad] -> logits (col 0) =====
        {
            v8f acc3 = vzero;
#pragma unroll
            for (int kt = 0; kt < 4; ++kt) {
                const v16h a = a_frag_f16(act + m * AST + kt * 32 + khalf * 8);
                const v16h b = b_frag(sW + HW_AW3 + ((size_t)kt * 32 + lane) * 16);
                acc3 = wmma16(a, b, acc3);
            }
            if (ncol == 0) {
                const float b3 = ab3[0];
#pragma unroll
                for (int v = 0; v < 8; ++v)
                    wsLog[tile * 16 + khalf * 8 + v] = (acc3[v] + b3) * 0.08838834764831845f;
            }
        }

        // ===== node_mlp layer 1: cat2[16,256] @ nw1[256,128], GELU =====
#pragma unroll
        for (int nt = 0; nt < 8; ++nt) acc[nt] = vzero;
#pragma unroll
        for (int kt = 0; kt < 8; ++kt) {
            const int c0 = kt * 32 + khalf * 8;
            const int f  = kt >> 2;                       // 0:e 1:hj
            const float* rp = (f == 0 ? ej : hj) + (c0 - f * 128);
            const v16h a = a_frag_f32(rp);
#pragma unroll
            for (int nt = 0; nt < 8; ++nt) {
                const v16h b = b_frag(sW + HW_NW1 + ((size_t)(kt * 8 + nt) * 32 + lane) * 16);
                acc[nt] = wmma16(a, b, acc[nt]);
            }
        }
#pragma unroll
        for (int nt = 0; nt < 8; ++nt) {
            const float bias = nb1[nt * 16 + ncol];
#pragma unroll
            for (int v = 0; v < 8; ++v) {
                const float x = gelu_exact(acc[nt][v] + bias);
                act[(khalf * 8 + v) * AST + nt * 16 + ncol] = (_Float16)x;
            }
        }
        lds_fence();

        // ===== node_mlp layer 2: [16,128] @ nw2[128,128], GELU =====
#pragma unroll
        for (int nt = 0; nt < 8; ++nt) acc[nt] = vzero;
#pragma unroll
        for (int kt = 0; kt < 4; ++kt) {
            const v16h a = a_frag_f16(act + m * AST + kt * 32 + khalf * 8);
#pragma unroll
            for (int nt = 0; nt < 8; ++nt) {
                const v16h b = b_frag(sW + HW_NW2 + ((size_t)(kt * 8 + nt) * 32 + lane) * 16);
                acc[nt] = wmma16(a, b, acc[nt]);
            }
        }
#pragma unroll
        for (int nt = 0; nt < 8; ++nt) {
            const float bias = nb2[nt * 16 + ncol];
#pragma unroll
            for (int v = 0; v < 8; ++v) {
                const float x = gelu_exact(acc[nt][v] + bias);
                act[(khalf * 8 + v) * AST + nt * 16 + ncol] = (_Float16)x;
            }
        }
        lds_fence();

        // ===== node_mlp layer 3: [16,128] @ nw3[128,128] + bias -> v =====
#pragma unroll
        for (int nt = 0; nt < 8; ++nt) acc[nt] = vzero;
#pragma unroll
        for (int kt = 0; kt < 4; ++kt) {
            const v16h a = a_frag_f16(act + m * AST + kt * 32 + khalf * 8);
#pragma unroll
            for (int nt = 0; nt < 8; ++nt) {
                const v16h b = b_frag(sW + HW_NW3 + ((size_t)(kt * 8 + nt) * 32 + lane) * 16);
                acc[nt] = wmma16(a, b, acc[nt]);
            }
        }
#pragma unroll
        for (int nt = 0; nt < 8; ++nt) {
            const float bias = nb3[nt * 16 + ncol];
#pragma unroll
            for (int v = 0; v < 8; ++v)
                act[(khalf * 8 + v) * AST + nt * 16 + ncol] = (_Float16)(acc[nt][v] + bias);
        }
        lds_fence();

        // coalesced copy of the [16 x 128] f16 value tile to scratch (128 B per lane)
        {
            const int row = lane >> 1;
            const int cb  = (lane & 1) * 64;
            _Float16* dp = wsV + (size_t)(tile * 16 + row) * DEMB + cb;
            const _Float16* sp = act + row * AST + cb;
#pragma unroll
            for (int c = 0; c < 64; c += 8)
                *(v8h*)(dp + c) = *(const v8h*)(sp + c);
        }
        lds_fence();   // act reads done before next iteration overwrites
    }
}

// ---------------- softmax over K=30 + attention-weighted aggregation ----------------
__global__ void __launch_bounds__(128)
pignn_agg_kernel(const float* __restrict__ wsLog, const _Float16* __restrict__ wsV,
                 _Float16* __restrict__ wsAgg) {
    __shared__ float sl[32];
    __shared__ float sw[32];
    const int node = blockIdx.x;
    const int t = threadIdx.x;

    if (t < 32) sl[t] = (t < KNBR) ? wsLog[(size_t)node * KNBR + t] : -1e30f;
    __syncthreads();

    float mx = -1e30f;
#pragma unroll
    for (int k = 0; k < KNBR; ++k) mx = fmaxf(mx, sl[k]);
    float ssum = 0.f;
#pragma unroll
    for (int k = 0; k < KNBR; ++k) ssum += __expf(sl[k] - mx);
    if (t < KNBR) sw[t] = __expf(sl[t] - mx) / ssum;
    __syncthreads();

    const _Float16* vp = wsV + (size_t)node * KNBR * DEMB + t;   // t == feature
    float a = 0.f;
#pragma unroll
    for (int k = 0; k < KNBR; ++k) a += sw[k] * (float)vp[k * DEMB];
    wsAgg[(size_t)node * DEMB + t] = (_Float16)a;
}

// ---------------- to_h GEMM (WMMA) + column-sum for scatter-mean ----------------
__global__ void __launch_bounds__(128)
pignn_toh_kernel(const _Float16* __restrict__ wsAgg, const _Float16* __restrict__ thwFrag,
                 float* __restrict__ wsHH, float* __restrict__ wsCol) {
    __shared__ float sCol[128];
    if (threadIdx.x < 128) sCol[threadIdx.x] = 0.f;
    __syncthreads();

    const int wave  = threadIdx.x >> 5;
    const int lane  = threadIdx.x & 31;
    const int m     = lane & 15;
    const int khalf = lane >> 4;
    const int ncol  = lane & 15;
    const int tile  = blockIdx.x * 4 + wave;

    const _Float16* row = wsAgg + (size_t)(tile * 16 + m) * DEMB;
    const v8f vzero = {0.f, 0.f, 0.f, 0.f, 0.f, 0.f, 0.f, 0.f};
    v8f acc[8];
#pragma unroll
    for (int nt = 0; nt < 8; ++nt) acc[nt] = vzero;
#pragma unroll
    for (int kt = 0; kt < 4; ++kt) {
        const v16h a = a_frag_f16(row + kt * 32 + khalf * 8);
#pragma unroll
        for (int nt = 0; nt < 8; ++nt) {
            const v16h b = b_frag(thwFrag + ((size_t)(kt * 8 + nt) * 32 + lane) * 16);
            acc[nt] = wmma16(a, b, acc[nt]);
        }
    }
#pragma unroll
    for (int nt = 0; nt < 8; ++nt) {
        float s = 0.f;
#pragma unroll
        for (int v = 0; v < 8; ++v) {
            const float val = acc[nt][v];
            wsHH[(size_t)(tile * 16 + khalf * 8 + v) * DEMB + nt * 16 + ncol] = val;
            s += val;
        }
        atomicAdd(&sCol[nt * 16 + ncol], s);
    }
    __syncthreads();
    if (threadIdx.x < 128) atomicAdd(&wsCol[threadIdx.x], sCol[threadIdx.x]);
}

// ---------------- gate MLP (single block, tiny) ----------------
__global__ void __launch_bounds__(128)
pignn_gate_kernel(const float* __restrict__ wsCol,
                  const float* __restrict__ gw1, const float* __restrict__ gb1,
                  const float* __restrict__ gw2, const float* __restrict__ gb2,
                  const float* __restrict__ gw3, const float* __restrict__ gb3,
                  float* __restrict__ wsG) {
    __shared__ float c0[128], c1[128], c2[128];
    const int t = threadIdx.x;
    c0[t] = wsCol[t] * (1.0f / (float)NNODE);
    __syncthreads();
    float s = gb1[t];
    for (int k = 0; k < 128; ++k) s += c0[k] * gw1[k * 128 + t];
    c1[t] = fmaxf(s, 0.f);
    __syncthreads();
    s = gb2[t];
    for (int k = 0; k < 128; ++k) s += c1[k] * gw2[k * 128 + t];
    c2[t] = fmaxf(s, 0.f);
    __syncthreads();
    s = gb3[t];
    for (int k = 0; k < 128; ++k) s += c2[k] * gw3[k * 128 + t];
    wsG[t] = 1.0f / (1.0f + __expf(-s));
}

// ---------------- final elementwise gating ----------------
__global__ void __launch_bounds__(256)
pignn_scale_kernel(const float* __restrict__ wsHH, const float* __restrict__ wsG,
                   float* __restrict__ out) {
    const size_t i = (size_t)blockIdx.x * blockDim.x + threadIdx.x;
    out[i] = wsHH[i] * wsG[i & 127];
}

// ---------------- launch ----------------
extern "C" void kernel_launch(void* const* d_in, const int* in_sizes, int n_in,
                              void* d_out, int out_size, void* d_ws, size_t ws_size,
                              hipStream_t stream) {
    const float* h   = (const float*)d_in[0];
    const float* e   = (const float*)d_in[1];
    const float* aw1 = (const float*)d_in[2];
    const float* ab1 = (const float*)d_in[3];
    const float* aw2 = (const float*)d_in[4];
    const float* ab2 = (const float*)d_in[5];
    const float* aw3 = (const float*)d_in[6];
    const float* ab3 = (const float*)d_in[7];
    const float* nw1 = (const float*)d_in[8];
    const float* nb1 = (const float*)d_in[9];
    const float* nw2 = (const float*)d_in[10];
    const float* nb2 = (const float*)d_in[11];
    const float* nw3 = (const float*)d_in[12];
    const float* nb3 = (const float*)d_in[13];
    const float* thw = (const float*)d_in[14];
    const float* gw1 = (const float*)d_in[15];
    const float* gb1 = (const float*)d_in[16];
    const float* gw2 = (const float*)d_in[17];
    const float* gb2 = (const float*)d_in[18];
    const float* gw3 = (const float*)d_in[19];
    const float* gb3 = (const float*)d_in[20];
    const int*  eidx = (const int*)d_in[21];

    char* ws = (char*)d_ws;
    _Float16* wsHalf = (_Float16*)ws;
    float*    wsLog  = (float*)(ws + WS_LOG);
    _Float16* wsV    = (_Float16*)(ws + WS_V);
    _Float16* wsAgg  = (_Float16*)(ws + WS_AGG);
    float*    wsHH   = (float*)(ws + WS_HH);
    float*    wsCol  = (float*)(ws + WS_COL);
    float*    wsG    = (float*)(ws + WS_G);

    const dim3 rb(256);
    pignn_repack_kernel<<<96, rb, 0, stream>>>(aw1, wsHalf + HW_AW1, 384, 128, 128);
    pignn_repack_kernel<<<32, rb, 0, stream>>>(aw2, wsHalf + HW_AW2, 128, 128, 128);
    pignn_repack_kernel<<<8,  rb, 0, stream>>>(aw3, wsHalf + HW_AW3, 128, 1,   16);
    pignn_repack_kernel<<<64, rb, 0, stream>>>(nw1, wsHalf + HW_NW1, 256, 128, 128);
    pignn_repack_kernel<<<32, rb, 0, stream>>>(nw2, wsHalf + HW_NW2, 128, 128, 128);
    pignn_repack_kernel<<<32, rb, 0, stream>>>(nw3, wsHalf + HW_NW3, 128, 128, 128);
    pignn_repack_kernel<<<32, rb, 0, stream>>>(thw, wsHalf + HW_THW, 128, 128, 128);
    pignn_zero_kernel<<<1, 128, 0, stream>>>(wsCol, 128);

    // 30720 tiles; 8 waves/block x 8 tiles/wave -> 480 blocks
    pignn_edge_kernel<<<NTILE / (WAVES * TPW), 256, 0, stream>>>(
        h, e, ab1, ab2, ab3, nb1, nb2, nb3, eidx, wsHalf, wsLog, wsV);
    pignn_agg_kernel<<<NNODE, 128, 0, stream>>>(wsLog, wsV, wsAgg);
    pignn_toh_kernel<<<NNODE / 64, 128, 0, stream>>>(wsAgg, wsHalf + HW_THW, wsHH, wsCol);
    pignn_gate_kernel<<<1, 128, 0, stream>>>(wsCol, gw1, gb1, gw2, gb2, gw3, gb3, wsG);
    pignn_scale_kernel<<<(NNODE * DEMB) / 256, 256, 0, stream>>>(wsHH, wsG, (float*)d_out);
}
